// LSTMLayer_6347961663606
// MI455X (gfx1250) — compile-verified, matrix operand order checked by reading, not run on hardware
//
#include <hip/hip_runtime.h>
#include <hip/hip_bf16.h>

// LSTM: T=1024, B=128, I=256, H=512. outputs [T,B,H] ++ h [B,H] ++ c [B,H] (f32).
//
// MI455X / gfx1250 strategy:
//  - bf16 WMMA (v_wmma_f32_16x16x32_bf16), f32 accumulate.
//  - Gate-interleaved weight packing: packed col q = j*4 + gate, so each WG
//    owns all 4 gates for 4 h-indices -> pointwise is WG-local (gates via LDS,
//    cell state c in registers for all 1024 steps).
//  - ONE device-wide barrier per step (publish h_t), split arrive/wait with
//    the next step's x-part GEMM (h-independent) overlapped in between.
//  - Weights resident in LDS across all steps; x stream prefetched 2 steps
//    ahead (global_prefetch_b8) off the critical path.

typedef __attribute__((ext_vector_type(16))) __bf16 bf16x16;
typedef __attribute__((ext_vector_type(8)))  __bf16 bf16x8;
typedef __attribute__((ext_vector_type(8)))  float  f32x8;
typedef __attribute__((ext_vector_type(4)))  float  f32x4;

#define TT   1024
#define BB   128
#define II   256
#define HH   512
#define G4H  2048         // 4*H packed gate columns
#define KTOT 768          // I + H
#define NWG  128          // persistent workgroups (= G4H/16)
#define LDB  776          // padded LDS weight row (768 + 8): 4-dword bank skew
#define GLD  20           // padded LDS gate row stride (floats)

__device__ __forceinline__ __bf16 f2bf(float f) {
  union { float f; unsigned u; } v; v.f = f;
  unsigned r = (v.u + 0x7FFFu + ((v.u >> 16) & 1u)) >> 16;   // RNE
  union { unsigned short s; __bf16 b; } o; o.s = (unsigned short)r;
  return o.b;
}

__device__ __forceinline__ float sigmoidf_fast(float x) {
  return 1.0f / (1.0f + __expf(-x));
}
__device__ __forceinline__ float tanhf_fast(float x) {
  return 1.0f - 2.0f / (__expf(2.0f * x) + 1.0f);
}

// ---------------- prep kernels ----------------

__global__ void prep_x(const float* __restrict__ x, __bf16* __restrict__ xb, long long n) {
  long long i = (long long)blockIdx.x * blockDim.x + threadIdx.x;
  if (i < n) xb[i] = f2bf(x[i]);
}

// Pack [W_ih | W_hh] rows, permuted: packed row q = j*4 + gate  <-  orig row gate*512 + j.
__global__ void prep_w(const float* __restrict__ Wih, const float* __restrict__ Whh,
                       const float* __restrict__ bih, const float* __restrict__ bhh,
                       __bf16* __restrict__ Wp, float* __restrict__ bias) {
  long long i = (long long)blockIdx.x * blockDim.x + threadIdx.x;
  if (i < (long long)G4H * KTOT) {
    int q = (int)(i / KTOT), k = (int)(i % KTOT);
    int gate = q & 3, j = q >> 2;
    int n_orig = gate * HH + j;
    float v = (k < II) ? Wih[(long long)n_orig * II + k]
                       : Whh[(long long)n_orig * HH + (k - II)];
    Wp[i] = f2bf(v);
  }
  if (i < G4H) {
    int q = (int)i;
    int n_orig = (q & 3) * HH + (q >> 2);
    bias[q] = bih[n_orig] + bhh[n_orig];
  }
}

__global__ void prep_state(const float* __restrict__ h0, __bf16* __restrict__ hb,
                           unsigned* __restrict__ cnt, unsigned* __restrict__ gen) {
  int i = blockIdx.x * blockDim.x + threadIdx.x;
  if (i < BB * HH) hb[i] = f2bf(h0[i]);
  if (i == 0) { *cnt = 0u; *gen = 0u; }
}

// ---------------- persistent LSTM kernel ----------------

__global__ __launch_bounds__(256, 1)
void lstm_persistent(const __bf16* __restrict__ xb,     // [T,B,I] bf16
                     const __bf16* __restrict__ Wp,     // [4H,768] bf16, gate-interleaved
                     const float*  __restrict__ bias,   // [4H] permuted b_ih+b_hh
                     const float*  __restrict__ c0,     // [B,H] initial cell
                     __bf16*       __restrict__ hb,     // [B,H] bf16 recurrent state
                     unsigned*     __restrict__ bar_cnt,
                     unsigned*     __restrict__ bar_gen,
                     float*        __restrict__ out)    // [T*B*H + 2*B*H]
{
  __shared__ __bf16 ldsB[16 * LDB];        // 24,832 B: this WG's 16 weight rows
  __shared__ float  gbuf[BB * GLD];        // 10,240 B: 128x16 gate tile (padded)

  const int tid   = threadIdx.x;
  const int wg    = blockIdx.x;            // owns packed cols q in [wg*16, wg*16+16)
  const int wave  = tid >> 5;              // M-tile: batch rows
  const int lane  = tid & 31;
  const int nloc  = lane & 15;
  const int khalf = (lane & 16) ? 8 : 0;   // 16-bit A/B wave32 K-half split
  const int n0    = wg << 4;
  const int m0    = wave << 4;
  const int mhi   = (lane >> 4) << 3;      // D layout: lanes 16-31 -> M=r+8
  const int mrow  = m0 + nloc;             // batch row held by this lane (A operand)

  // Pointwise ownership: cell e covers (b, j) with j = wg*4 + jl.
  int pw_b[2], pw_jl[2];
  float creg[2];
  #pragma unroll
  for (int e = 0; e < 2; ++e) {
    int cell = e * 256 + tid;              // 0..511
    pw_jl[e] = cell & 3;
    pw_b[e]  = cell >> 2;
    creg[e]  = c0[(long long)pw_b[e] * HH + (wg << 2) + pw_jl[e]];
  }

  // Stage weights into padded LDS once; reused for all T steps.
  for (int idx = tid; idx < 16 * KTOT; idx += 256) {
    int r = idx / KTOT, k = idx % KTOT;
    ldsB[r * LDB + k] = Wp[(long long)(n0 + r) * KTOT + k];
  }
  const float bn = bias[n0 + nloc];
  __syncthreads();

  // x-part of step 0 (h-independent).
  f32x8 acc = {};
  {
    const __bf16* xrow = xb + (long long)mrow * II;   // t = 0
    #pragma unroll
    for (int kk = 0; kk < II; kk += 32) {
      union { bf16x16 f; struct { bf16x8 lo, hi; } h; } a, b;
      a.h.lo = *(const bf16x8*)(xrow + kk + khalf);
      a.h.hi = *(const bf16x8*)(xrow + kk + 16 + khalf);
      b.h.lo = *(const bf16x8*)(&ldsB[nloc * LDB + kk + khalf]);
      b.h.hi = *(const bf16x8*)(&ldsB[nloc * LDB + kk + 16 + khalf]);
      acc = __builtin_amdgcn_wmma_f32_16x16x32_bf16(false, a.f, false, b.f,
                                                    (short)0, acc, false, false);
    }
  }

  const long long TBH = (long long)TT * BB * HH;

  for (int t = 0; t < TT; ++t) {
    // ---- h-part GEMM: K = 512 over h_{t-1} ----
    {
      const __bf16* hrow = hb + (long long)mrow * HH;
      #pragma unroll
      for (int kk = 0; kk < HH; kk += 32) {
        union { bf16x16 f; struct { bf16x8 lo, hi; } h; } a, b;
        a.h.lo = *(const bf16x8*)(hrow + kk + khalf);
        a.h.hi = *(const bf16x8*)(hrow + kk + 16 + khalf);
        b.h.lo = *(const bf16x8*)(&ldsB[nloc * LDB + II + kk + khalf]);
        b.h.hi = *(const bf16x8*)(&ldsB[nloc * LDB + II + kk + 16 + khalf]);
        acc = __builtin_amdgcn_wmma_f32_16x16x32_bf16(false, a.f, false, b.f,
                                                      (short)0, acc, false, false);
      }
    }

    // ---- gate tile (+bias) -> LDS (WG-local; never touches global) ----
    #pragma unroll
    for (int r = 0; r < 8; ++r)
      gbuf[(m0 + mhi + r) * GLD + nloc] = acc[r] + bn;
    __syncthreads();

    // ---- pointwise: c in registers, publish h_t ----
    #pragma unroll
    for (int e = 0; e < 2; ++e) {
      const int b = pw_b[e], jl = pw_jl[e];
      const int j = (wg << 2) + jl;
      f32x4 g = *(const f32x4*)(&gbuf[b * GLD + (jl << 2)]);   // i,f,g,o
      float ig = sigmoidf_fast(g.x);
      float fg = sigmoidf_fast(g.y);
      float gg = tanhf_fast(g.z);
      float og = sigmoidf_fast(g.w);
      float cn = fg * creg[e] + ig * gg;
      float hn = og * tanhf_fast(cn);
      creg[e] = cn;
      const long long p = (long long)b * HH + j;
      hb[p] = f2bf(hn);
      out[(long long)t * (BB * HH) + p] = hn;
      if (t == TT - 1) out[TBH + p] = hn;                      // final h
    }

    // ---- grid barrier: arrive, overlap next x-part + prefetch, wait ----
    __threadfence();                    // make this thread's hb stores visible
    __syncthreads();                    // whole WG arrived
    unsigned g_obs = 0u;
    if (tid == 0) {
      g_obs = __atomic_load_n(bar_gen, __ATOMIC_ACQUIRE);
      if (atomicAdd(bar_cnt, 1u) == (unsigned)(NWG - 1)) {
        __atomic_store_n(bar_cnt, 0u, __ATOMIC_RELAXED);
        __threadfence();
        __atomic_fetch_add(bar_gen, 1u, __ATOMIC_RELEASE);
      }
    }

    // Overlap: x-part GEMM of step t+1 (independent of h_t).
    if (t + 1 < TT) {
      acc = (f32x8){};
      const __bf16* xrow = xb + ((long long)(t + 1) * BB + mrow) * II;
      #pragma unroll
      for (int kk = 0; kk < II; kk += 32) {
        union { bf16x16 f; struct { bf16x8 lo, hi; } h; } a, b;
        a.h.lo = *(const bf16x8*)(xrow + kk + khalf);
        a.h.hi = *(const bf16x8*)(xrow + kk + 16 + khalf);
        b.h.lo = *(const bf16x8*)(&ldsB[nloc * LDB + kk + khalf]);
        b.h.hi = *(const bf16x8*)(&ldsB[nloc * LDB + kk + 16 + khalf]);
        acc = __builtin_amdgcn_wmma_f32_16x16x32_bf16(false, a.f, false, b.f,
                                                      (short)0, acc, false, false);
      }
    }

    // Prefetch this lane's x row for step t+2 (global_prefetch_b8): pulls the
    // next slab of the 67 MB x stream toward L2/WGP$ off the critical path.
    if (t + 2 < TT) {
      const char* pf = (const char*)(xb + ((long long)(t + 2) * BB + mrow) * II);
      __builtin_prefetch(pf, 0, 1);
      __builtin_prefetch(pf + 256, 0, 1);   // row is 512 B
    }

    if (tid == 0) {
      while (__atomic_load_n(bar_gen, __ATOMIC_ACQUIRE) == g_obs)
        __builtin_amdgcn_s_sleep(2);
    }
    __syncthreads();
  }

  // ---- final c from registers ----
  #pragma unroll
  for (int e = 0; e < 2; ++e) {
    const long long p = (long long)pw_b[e] * HH + (wg << 2) + pw_jl[e];
    out[TBH + (long long)BB * HH + p] = creg[e];
  }
}

// ---------------- launch ----------------

extern "C" void kernel_launch(void* const* d_in, const int* in_sizes, int n_in,
                              void* d_out, int out_size, void* d_ws, size_t ws_size,
                              hipStream_t stream) {
  const float* x    = (const float*)d_in[0];
  const float* h0   = (const float*)d_in[1];
  const float* c0   = (const float*)d_in[2];
  const float* Wih  = (const float*)d_in[3];
  const float* Whh  = (const float*)d_in[4];
  const float* bih  = (const float*)d_in[5];
  const float* bhh  = (const float*)d_in[6];
  float* out = (float*)d_out;

  // workspace layout (256B-aligned regions, ~70.4 MB total)
  char* ws = (char*)d_ws;
  size_t off = 0;
  __bf16*   xb   = (__bf16*)(ws + off); off += (size_t)TT * BB * II * 2;   // 67,108,864
  __bf16*   Wp   = (__bf16*)(ws + off); off += (size_t)G4H * KTOT * 2;     //  3,145,728
  float*    bias = (float*)(ws + off);  off += (size_t)G4H * 4;            //      8,192
  __bf16*   hb   = (__bf16*)(ws + off); off += (size_t)BB * HH * 2;        //    131,072
  unsigned* bcnt = (unsigned*)(ws + off);
  unsigned* bgen = bcnt + 1;

  const long long nx = (long long)TT * BB * II;
  prep_x<<<(unsigned)((nx + 255) / 256), 256, 0, stream>>>(x, xb, nx);

  const long long nw = (long long)G4H * KTOT;
  prep_w<<<(unsigned)((nw + 255) / 256), 256, 0, stream>>>(Wih, Whh, bih, bhh, Wp, bias);

  prep_state<<<(BB * HH + 255) / 256, 256, 0, stream>>>(h0, hb, bcnt, bgen);

  lstm_persistent<<<NWG, 256, 0, stream>>>(xb, Wp, bias, c0, hb, bcnt, bgen, out);
}
